// SparseLatentMoE_42726334660621
// MI455X (gfx1250) — matrix operand order repaired and checked by reading, not compiled, not run on hardware
//
#include <hip/hip_runtime.h>
#include <hip/hip_bf16.h>
#include <math.h>

// ---------------------------------------------------------------------------
// Types
// ---------------------------------------------------------------------------
typedef _Float16 h16;
typedef __attribute__((ext_vector_type(8)))  _Float16 v8h;
typedef __attribute__((ext_vector_type(16))) _Float16 v16h;
typedef __attribute__((ext_vector_type(8)))  float    v8f;
typedef __attribute__((ext_vector_type(4)))  float    v4f;

// Problem dims: B=64 T=512 D=512 S=128 E=64 H=4 HD=128, TOPK_SLOTS=8, TOPK_EXPERTS=2

// d_out layout (floats), reference tuple return order
static constexpr size_t OFF_OUT  = 0;                       // [64,512]
static constexpr size_t OFF_NS   = 32768;                   // [64,65536]
static constexpr size_t OFF_SIDX = 32768 + 4194304;         // [64,8]
static constexpr size_t OFF_RIDX = OFF_SIDX + 512;          // [64,8]
static constexpr size_t OFF_EIDX = OFF_RIDX + 512;          // [64,2]
static constexpr size_t OFF_SW   = OFF_EIDX + 128;          // [64,8]
static constexpr size_t OFF_EW   = OFF_SW + 512;            // [64,2]

static __device__ __forceinline__ v16h cat8(v8h a, v8h b) {
  return __builtin_shufflevector(a, b, 0,1,2,3,4,5,6,7,8,9,10,11,12,13,14,15);
}
static __device__ __forceinline__ v8f wmma16(v16h a, v16h b, v8f c) {
  // D = A(16x32 f16) * B(32x16 f16) + C(16x16 f32)
  return __builtin_amdgcn_wmma_f32_16x16x32_f16(false, a, false, b, (short)0, c,
                                                false, false);
}
static __device__ __forceinline__ v8f zero8f() {
  v8f z;
#pragma unroll
  for (int i = 0; i < 8; ++i) z[i] = 0.f;
  return z;
}

// ---------------------------------------------------------------------------
// CDNA5 async global->LDS copies (ASYNCcnt path, ISA 08_async_tensor 4.x).
// INST_OFFSET is added to BOTH the LDS (VDST VGPR) and global addresses, so a
// contiguous run needs one LDS-address VGPR + one 64-bit global address.
// Generic pointers to __shared__ carry the LDS offset in their low 32 bits
// (flat->LDS mapping uses addr[31:0], ISA 10.2), so truncation is the LDS addr.
// ---------------------------------------------------------------------------
static __device__ __forceinline__ void async_copy32B(void* lds_dst, const void* gsrc) {
  unsigned l0 = (unsigned)(size_t)lds_dst;
  asm volatile(
      "global_load_async_to_lds_b128 %0, %1, off\n\t"
      "global_load_async_to_lds_b128 %0, %1, off offset:16"
      :: "v"(l0), "v"(gsrc) : "memory");
}
static __device__ __forceinline__ void async_copy64B(void* lds_dst, const void* gsrc) {
  unsigned l0 = (unsigned)(size_t)lds_dst;
  asm volatile(
      "global_load_async_to_lds_b128 %0, %1, off\n\t"
      "global_load_async_to_lds_b128 %0, %1, off offset:16\n\t"
      "global_load_async_to_lds_b128 %0, %1, off offset:32\n\t"
      "global_load_async_to_lds_b128 %0, %1, off offset:48"
      :: "v"(l0), "v"(gsrc) : "memory");
}
static __device__ __forceinline__ void wait_async0() {
  asm volatile("s_wait_asynccnt 0" ::: "memory");
}

// ---------------------------------------------------------------------------
// Generic WMMA GEMM:  C[M,N] = rowmap(A)[M,K] @ W[N,K]^T + bias[N]
// A is f32 or f16; W/bias f32; C f32 or f16. Optional dual A source:
// global row g -> b=g/group, r=g%group; r<split from A0 else A1 (for kv concat)
// Block: 128x128 macro tile, 256 threads = 8 waves (4 M x 2 N), BK=32.
// f16 A tiles are DMA'd to LDS with global_load_async_to_lds_b128.
// ---------------------------------------------------------------------------
template <typename TA, typename TC>
__global__ __launch_bounds__(256)
void gemm_f16_wmma(const TA* __restrict__ A0, const TA* __restrict__ A1,
                   int group, int split,
                   const float* __restrict__ W, const float* __restrict__ bias,
                   TC* __restrict__ C, int M, int N, int K) {
  __shared__ __align__(16) h16 As[128][40];  // [m][k] f16 (+pad)
  __shared__ __align__(16) h16 Bs[128][40];  // [n][k] f16 == B^T tile (+pad)
  const int tid = threadIdx.x;
  const int wave = tid >> 5, lane = tid & 31;
  const int wm = wave >> 1, wn = wave & 1;
  const int lo = lane & 15, hi = lane >> 4;
  const int bm0 = blockIdx.y * 128, bn0 = blockIdx.x * 128;

  v8f acc[2][4];
#pragma unroll
  for (int i = 0; i < 2; ++i)
#pragma unroll
    for (int j = 0; j < 4; ++j) acc[i][j] = zero8f();

  // loader assignment: thread -> (row lr, 16-wide k segment lc)
  const int lr = tid >> 1;
  const int lc = (tid & 1) << 4;

  const TA* arow = nullptr;
  {
    int gm = bm0 + lr;
    if (gm < M) {
      if (A1) {
        int bb = gm / group;
        int rr = gm - bb * group;
        arow = (rr < split) ? (A0 + (size_t)(bb * split + rr) * K)
                            : (A1 + (size_t)(bb * (group - split) + (rr - split)) * K);
      } else {
        arow = A0 + (size_t)gm * K;
      }
    }
  }
  const float* wrow = nullptr;
  {
    int gn = bn0 + lr;
    if (gn < N) wrow = W + (size_t)gn * K;
  }

  for (int k0 = 0; k0 < K; k0 += 32) {
    __syncthreads();
    {  // A tile -> f16 LDS
      h16* dst = &As[lr][lc];
      if (arow) {
        if constexpr (sizeof(TA) == 4) {
          if (k0 + 64 < K) __builtin_prefetch(arow + k0 + 64, 0, 1);
#pragma unroll
          for (int u = 0; u < 16; u += 4) {
            v4f t = *(const v4f*)(arow + k0 + lc + u);
            dst[u + 0] = (h16)t.x; dst[u + 1] = (h16)t.y;
            dst[u + 2] = (h16)t.z; dst[u + 3] = (h16)t.w;
          }
        } else {
          // raw f16 bytes: async DMA straight into LDS (no VGPR staging)
          async_copy32B(dst, arow + k0 + lc);
        }
      } else {
#pragma unroll
        for (int u = 0; u < 16; ++u) dst[u] = (h16)0.f;
      }
    }
    {  // W tile -> f16 LDS (W row-major [N,K] is exactly B^T)
      h16* dst = &Bs[lr][lc];
      if (wrow) {
        if (k0 + 64 < K) __builtin_prefetch(wrow + k0 + 64, 0, 1);
#pragma unroll
        for (int u = 0; u < 16; u += 4) {
          v4f t = *(const v4f*)(wrow + k0 + lc + u);
          dst[u + 0] = (h16)t.x; dst[u + 1] = (h16)t.y;
          dst[u + 2] = (h16)t.z; dst[u + 3] = (h16)t.w;
        }
      } else {
#pragma unroll
        for (int u = 0; u < 16; ++u) dst[u] = (h16)0.f;
      }
    }
    if constexpr (sizeof(TA) == 2) wait_async0();  // wave's async DMAs done
    __syncthreads();

    // ISA A-layout (16-bit, 16x32): lane m=lo, lo8 halves at k=hi*8, hi8 at 16+hi*8
    v16h af[2];
#pragma unroll
    for (int mt = 0; mt < 2; ++mt) {
      int m = wm * 32 + mt * 16 + lo;
      af[mt] = cat8(*(const v8h*)&As[m][hi * 8], *(const v8h*)&As[m][16 + hi * 8]);
    }
    // ISA B-layout: lane n=lo, lo8 halves at k=hi*8, hi8 at 16+hi*8 (same as A)
#pragma unroll
    for (int nt = 0; nt < 4; ++nt) {
      int n = wn * 64 + nt * 16 + lo;
      v16h bf = cat8(*(const v8h*)&Bs[n][hi * 8], *(const v8h*)&Bs[n][16 + hi * 8]);
      acc[0][nt] = wmma16(af[0], bf, acc[0][nt]);
      acc[1][nt] = wmma16(af[1], bf, acc[1][nt]);
    }
  }

  // C-layout: row = i + hi*8, col = lo
#pragma unroll
  for (int nt = 0; nt < 4; ++nt) {
    int col = bn0 + wn * 64 + nt * 16 + lo;
    if (col >= N) continue;
    float bv = bias ? bias[col] : 0.f;
#pragma unroll
    for (int mt = 0; mt < 2; ++mt) {
#pragma unroll
      for (int i = 0; i < 8; ++i) {
        int row = bm0 + wm * 32 + mt * 16 + hi * 8 + i;
        if (row < M) C[(size_t)row * N + col] = (TC)(acc[mt][nt][i] + bv);
      }
    }
  }
}

// ---------------------------------------------------------------------------
// Flash attention per (b,h): Q[128x128] vs KV[640x128], online softmax,
// both GEMMs in WMMA f16. Block = 8 waves, wave w owns query rows 16w..16w+15.
// K chunks are DMA'd to LDS with async-to-LDS; V is transposed through VGPRs.
// ---------------------------------------------------------------------------
__global__ __launch_bounds__(256)
void flash_attn_kernel(const h16* __restrict__ qg, const h16* __restrict__ kg,
                       const h16* __restrict__ vg, h16* __restrict__ og) {
  const int b = blockIdx.x >> 2, h = blockIdx.x & 3;
  const int tid = threadIdx.x, wave = tid >> 5, lane = tid & 31;
  const int lo = lane & 15, hi = lane >> 4;
  __shared__ __align__(16) h16 KV[9216];      // Kt[64][136] or Vt[128][72]
  __shared__ __align__(16) h16 Ps[128 * 72];  // P[128][72] f16

  // Preload Q A-fragments straight from global (A-layout addresses are 16B runs)
  v16h aq[4];
  {
    const h16* qr = qg + ((size_t)(b * 128 + wave * 16 + lo)) * 512 + h * 128;
#pragma unroll
    for (int ks = 0; ks < 4; ++ks)
      aq[ks] = cat8(*(const v8h*)(qr + ks * 32 + hi * 8),
                    *(const v8h*)(qr + ks * 32 + 16 + hi * 8));
  }

  v8f oacc[8];
#pragma unroll
  for (int t = 0; t < 8; ++t) oacc[t] = zero8f();
  float mrow[8], lrow[8];
#pragma unroll
  for (int i = 0; i < 8; ++i) { mrow[i] = -INFINITY; lrow[i] = 0.f; }
  const float scale = 0.0883883476483184f;  // 1/sqrt(128)

  for (int c = 0; c < 10; ++c) {  // 640 keys in chunks of 64
    __syncthreads();
    {  // K chunk -> LDS [key][hd] via async DMA (raw f16, no conversion)
      int j = tid >> 2, seg = (tid & 3) << 5;
      const h16* src = kg + ((size_t)(b * 640 + c * 64 + j)) * 512 + h * 128 + seg;
      async_copy64B(&KV[j * 136 + seg], src);
    }
    wait_async0();
    __syncthreads();

    v8f sacc[4];
#pragma unroll
    for (int nt = 0; nt < 4; ++nt) sacc[nt] = zero8f();
#pragma unroll
    for (int ks = 0; ks < 4; ++ks) {
#pragma unroll
      for (int nt = 0; nt < 4; ++nt) {
        int key = nt * 16 + lo;
        v16h bf = cat8(*(const v8h*)&KV[key * 136 + ks * 32 + hi * 8],
                       *(const v8h*)&KV[key * 136 + ks * 32 + 16 + hi * 8]);
        sacc[nt] = wmma16(aq[ks], bf, sacc[nt]);
      }
    }
#pragma unroll
    for (int nt = 0; nt < 4; ++nt)
#pragma unroll
      for (int i = 0; i < 8; ++i) sacc[nt][i] *= scale;

    // row max: reduce across the 16 lanes sharing a row (xor bits 0..3)
    float mloc[8];
#pragma unroll
    for (int i = 0; i < 8; ++i) {
      float v = fmaxf(fmaxf(sacc[0][i], sacc[1][i]), fmaxf(sacc[2][i], sacc[3][i]));
      v = fmaxf(v, __shfl_xor(v, 1, 32));
      v = fmaxf(v, __shfl_xor(v, 2, 32));
      v = fmaxf(v, __shfl_xor(v, 4, 32));
      v = fmaxf(v, __shfl_xor(v, 8, 32));
      mloc[i] = v;
    }
    float corr[8], psum[8];
#pragma unroll
    for (int i = 0; i < 8; ++i) {
      float mn = fmaxf(mrow[i], mloc[i]);
      corr[i] = __expf(mrow[i] - mn);
      mrow[i] = mn;
      psum[i] = 0.f;
    }
    // P = exp(s - m) -> f16 LDS (re-layout C->A for the PV GEMM)
#pragma unroll
    for (int nt = 0; nt < 4; ++nt) {
#pragma unroll
      for (int i = 0; i < 8; ++i) {
        float p = __expf(sacc[nt][i] - mrow[i]);
        psum[i] += p;
        Ps[(wave * 16 + hi * 8 + i) * 72 + nt * 16 + lo] = (h16)p;
      }
    }
#pragma unroll
    for (int i = 0; i < 8; ++i) {
      float s = psum[i];
      s += __shfl_xor(s, 1, 32);
      s += __shfl_xor(s, 2, 32);
      s += __shfl_xor(s, 4, 32);
      s += __shfl_xor(s, 8, 32);
      lrow[i] = lrow[i] * corr[i] + s;
    }
#pragma unroll
    for (int t = 0; t < 8; ++t)
#pragma unroll
      for (int i = 0; i < 8; ++i) oacc[t][i] *= corr[i];

    __syncthreads();  // all waves done reading Kt before overwriting with Vt
    {  // V chunk -> LDS transposed Vt[hd][key] (B^T for P@V)
      int j = tid & 63, hb = (tid >> 6) << 5;
      const h16* src = vg + ((size_t)(b * 640 + c * 64 + j)) * 512 + h * 128 + hb;
#pragma unroll
      for (int u = 0; u < 32; u += 8) {
        v8h tv = *(const v8h*)(src + u);
#pragma unroll
        for (int e = 0; e < 8; ++e) KV[(hb + u + e) * 72 + j] = tv[e];
      }
    }
    __syncthreads();

    // O += P[16x64] @ V[64x128]
#pragma unroll
    for (int ks2 = 0; ks2 < 2; ++ks2) {
      v16h af = cat8(*(const v8h*)&Ps[(wave * 16 + lo) * 72 + ks2 * 32 + hi * 8],
                     *(const v8h*)&Ps[(wave * 16 + lo) * 72 + ks2 * 32 + 16 + hi * 8]);
#pragma unroll
      for (int t = 0; t < 8; ++t) {
        v16h bf = cat8(*(const v8h*)&KV[(t * 16 + lo) * 72 + ks2 * 32 + hi * 8],
                       *(const v8h*)&KV[(t * 16 + lo) * 72 + ks2 * 32 + 16 + hi * 8]);
        oacc[t] = wmma16(af, bf, oacc[t]);
      }
    }
  }

#pragma unroll
  for (int t = 0; t < 8; ++t) {
    int col = h * 128 + t * 16 + lo;
#pragma unroll
    for (int i = 0; i < 8; ++i) {
      int row = b * 128 + wave * 16 + hi * 8 + i;
      og[(size_t)row * 512 + col] = (h16)(oacc[t][i] / lrow[i]);
    }
  }
}

// ---------------------------------------------------------------------------
// Small kernels
// ---------------------------------------------------------------------------
static __device__ __forceinline__ float blk_sum(float v, float* sb) {
#pragma unroll
  for (int m = 16; m >= 1; m >>= 1) v += __shfl_xor(v, m, 32);
  __syncthreads();
  if ((threadIdx.x & 31) == 0) sb[threadIdx.x >> 5] = v;
  __syncthreads();
  float r = 0.f;
#pragma unroll
  for (int i = 0; i < 8; ++i) r += sb[i];
  return r;
}

// LayerNorm on state slots (-> f16) + read scores on RAW slots
__global__ __launch_bounds__(256)
void ln_read_kernel(const float* __restrict__ state, const float* __restrict__ g,
                    const float* __restrict__ beta, const float* __restrict__ rw,
                    const float* __restrict__ rb, h16* __restrict__ sn,
                    float* __restrict__ rscore) {
  __shared__ float sb[8];
  const int slot = blockIdx.x;  // 0..B*S-1
  const int t = threadIdx.x;
  const float* xs = state + (size_t)slot * 512;
  float x0 = xs[t], x1 = xs[t + 256];
  float s  = blk_sum(x0 + x1, sb);
  float s2 = blk_sum(x0 * x0 + x1 * x1, sb);
  float rd = blk_sum(x0 * rw[t] + x1 * rw[t + 256], sb);
  float mu = s * (1.f / 512.f);
  float var = s2 * (1.f / 512.f) - mu * mu;
  float rs = rsqrtf(var + 1e-5f);
  h16* o = sn + (size_t)slot * 512;
  o[t]       = (h16)((x0 - mu) * rs * g[t] + beta[t]);
  o[t + 256] = (h16)((x1 - mu) * rs * g[t + 256] + beta[t + 256]);
  if (t == 0) rscore[slot] = rd + rb[0];
}

// gate on mean(attn_out over slots) + top2 + softmax.
// Note: expert_weights sum to 1, so the expert_out scale is exactly 1.
__global__ __launch_bounds__(256)
void gate_topk_kernel(const float* __restrict__ ao, const float* __restrict__ gw,
                      const float* __restrict__ gb, float* __restrict__ dout) {
  const int b = blockIdx.x, t = threadIdx.x;
  __shared__ float mv[512];
  __shared__ float gl[64];
  for (int d = t; d < 512; d += 256) {
    float s = 0.f;
    const float* p = ao + ((size_t)b * 128) * 512 + d;
    for (int sl = 0; sl < 128; ++sl) s += p[(size_t)sl * 512];
    mv[d] = s * (1.f / 128.f);
  }
  __syncthreads();
  if (t < 64) {
    float a = gb[t];
    const float* wr = gw + (size_t)t * 512;
    for (int d = 0; d < 512; ++d) a += mv[d] * wr[d];
    gl[t] = a;
  }
  __syncthreads();
  if (t == 0) {
    int i0 = 0; float v0 = gl[0];
    for (int e = 1; e < 64; ++e) if (gl[e] > v0) { v0 = gl[e]; i0 = e; }
    int i1 = (i0 == 0) ? 1 : 0; float v1 = -INFINITY;
    for (int e = 0; e < 64; ++e) if (e != i0 && gl[e] > v1) { v1 = gl[e]; i1 = e; }
    float w1 = __expf(v1 - v0);
    float inv = 1.f / (1.f + w1);
    dout[OFF_EIDX + b * 2 + 0] = (float)i0;
    dout[OFF_EIDX + b * 2 + 1] = (float)i1;
    dout[OFF_EW + b * 2 + 0] = inv;
    dout[OFF_EW + b * 2 + 1] = w1 * inv;
  }
}

__global__ __launch_bounds__(256)
void slot_topk_kernel(const float* __restrict__ eo, const float* __restrict__ sw,
                      const float* __restrict__ sbias, const float* __restrict__ rscore,
                      int* __restrict__ sidx, float* __restrict__ dout) {
  const int b = blockIdx.x, t = threadIdx.x;
  __shared__ float sc[128];
  if (t < 128) {
    const float* r = eo + ((size_t)(b * 128 + t)) * 512;
    float a = sbias[0];
    for (int d = 0; d < 512; ++d) a += r[d] * sw[d];
    sc[t] = a;
  }
  __syncthreads();
  if (t == 0) {  // top8 slots (strict > scan == lax.top_k tie-break)
    int idx[8]; float val[8];
    unsigned used[4] = {0, 0, 0, 0};
    for (int j = 0; j < 8; ++j) {
      float best = -INFINITY; int bi = 0;
      for (int s = 0; s < 128; ++s) {
        if (used[s >> 5] & (1u << (s & 31))) continue;
        if (sc[s] > best) { best = sc[s]; bi = s; }
      }
      used[bi >> 5] |= 1u << (bi & 31);
      idx[j] = bi; val[j] = best;
    }
    float m = val[0], sum = 0.f, w[8];
    for (int j = 0; j < 8; ++j) { w[j] = __expf(val[j] - m); sum += w[j]; }
    float inv = 1.f / sum;
    for (int j = 0; j < 8; ++j) {
      dout[OFF_SIDX + b * 8 + j] = (float)idx[j];
      dout[OFF_SW + b * 8 + j] = w[j] * inv;
      sidx[b * 8 + j] = idx[j];
    }
  }
  if (t == 1) {  // top8 read indices
    const float* rs = rscore + b * 128;
    unsigned used[4] = {0, 0, 0, 0};
    for (int j = 0; j < 8; ++j) {
      float best = -INFINITY; int bi = 0;
      for (int s = 0; s < 128; ++s) {
        if (used[s >> 5] & (1u << (s & 31))) continue;
        if (rs[s] > best) { best = rs[s]; bi = s; }
      }
      used[bi >> 5] |= 1u << (bi & 31);
      dout[OFF_RIDX + b * 8 + j] = (float)bi;
    }
  }
}

__global__ __launch_bounds__(256)
void gather_kernel(const float* __restrict__ eo, const float* __restrict__ state,
                   const int* __restrict__ sidx, float* __restrict__ geo,
                   float* __restrict__ gst) {
  const int b = blockIdx.x;
  for (int t = threadIdx.x; t < 8 * 512; t += 256) {
    int j = t >> 9, d = t & 511;
    int row = sidx[b * 8 + j];
    geo[((size_t)(b * 8 + j)) * 512 + d] = eo[((size_t)(b * 128 + row)) * 512 + d];
    gst[((size_t)(b * 8 + j)) * 512 + d] = state[(size_t)b * 65536 + (size_t)row * 512 + d];
  }
}

__global__ __launch_bounds__(256)
void copy_state_kernel(const float* __restrict__ state, float* __restrict__ ns) {
  size_t i = (size_t)blockIdx.x * 256 + threadIdx.x;  // in float4s
  ((float4*)ns)[i] = ((const float4*)state)[i];
}

__global__ __launch_bounds__(256)
void scatter_kernel(const float* __restrict__ gst, const float* __restrict__ upd,
                    const int* __restrict__ sidx, const float* __restrict__ swgt,
                    float* __restrict__ ns) {
  const int b = blockIdx.x;
  for (int t = threadIdx.x; t < 8 * 512; t += 256) {
    int j = t >> 9, d = t & 511;
    int row = sidx[b * 8 + j];
    float w = swgt[b * 8 + j];
    float u = tanhf(upd[((size_t)(b * 8 + j)) * 512 + d]);
    ns[(size_t)b * 65536 + (size_t)row * 512 + d] =
        0.7f * gst[((size_t)(b * 8 + j)) * 512 + d] + 0.3f * w * u;
  }
}

// ---------------------------------------------------------------------------
// Host launcher
// ---------------------------------------------------------------------------
extern "C" void kernel_launch(void* const* d_in, const int* in_sizes, int n_in,
                              void* d_out, int out_size, void* d_ws, size_t ws_size,
                              hipStream_t stream) {
  (void)in_sizes; (void)n_in; (void)out_size; (void)ws_size;
  const float* x   = (const float*)d_in[0];
  const float* st  = (const float*)d_in[1];
  const float* inW = (const float*)d_in[2];
  const float* inB = (const float*)d_in[3];
  const float* lng = (const float*)d_in[4];
  const float* lnb = (const float*)d_in[5];
  const float* aiW = (const float*)d_in[6];
  const float* aiB = (const float*)d_in[7];
  const float* aoW = (const float*)d_in[8];
  const float* aoB = (const float*)d_in[9];
  const float* exW = (const float*)d_in[10];
  const float* exB = (const float*)d_in[11];
  const float* gW  = (const float*)d_in[12];
  const float* gB  = (const float*)d_in[13];
  const float* rW  = (const float*)d_in[14];
  const float* rB  = (const float*)d_in[15];
  const float* sgW = (const float*)d_in[16];
  const float* sgB = (const float*)d_in[17];
  const float* stW = (const float*)d_in[18];
  const float* stB = (const float*)d_in[19];
  const float* oW  = (const float*)d_in[20];
  const float* oB  = (const float*)d_in[21];
  float* out = (float*)d_out;

  char* ws = (char*)d_ws;
  size_t off = 0;
  auto take = [&](size_t bytes) -> char* {
    off = (off + 255) & ~(size_t)255;
    char* p = ws + off;
    off += bytes;
    return p;
  };
  h16*   xp   = (h16*)take((size_t)32768 * 512 * 2);  // x_proj (f16)
  h16*   snb  = (h16*)take((size_t)8192 * 512 * 2);   // slots_norm (f16)
  h16*   qb   = (h16*)take((size_t)8192 * 512 * 2);   // q (f16)
  h16*   kb   = (h16*)take((size_t)40960 * 512 * 2);  // k (f16)
  h16*   vb   = (h16*)take((size_t)40960 * 512 * 2);  // v (f16)
  h16*   atn  = (h16*)take((size_t)8192 * 512 * 2);   // attn (f16)
  float* aob  = (float*)take((size_t)8192 * 512 * 4); // attn_out (f32)
  float* eob  = (float*)take((size_t)8192 * 512 * 4); // expert_out (f32)
  float* rsc  = (float*)take((size_t)8192 * 4);       // read scores
  float* geo  = (float*)take((size_t)512 * 512 * 4);  // gathered expert_out
  float* gst  = (float*)take((size_t)512 * 512 * 4);  // gathered state
  float* updb = (float*)take((size_t)512 * 512 * 4);  // state MLP pre-tanh
  int*   sidx = (int*)take((size_t)512 * 4);          // slot indices

  float* ns = out + OFF_NS;

  // 1. LayerNorm slots + read scores
  ln_read_kernel<<<8192, 256, 0, stream>>>(st, lng, lnb, rW, rB, snb, rsc);
  // 2. x_proj = x @ in_proj^T + b
  gemm_f16_wmma<float, h16><<<dim3(4, 256), 256, 0, stream>>>(
      x, (const float*)nullptr, 0, 0, inW, inB, xp, 32768, 512, 512);
  // 3. q = slots_norm @ wq^T + bq
  gemm_f16_wmma<h16, h16><<<dim3(4, 64), 256, 0, stream>>>(
      snb, (const h16*)nullptr, 0, 0, aiW, aiB, qb, 8192, 512, 512);
  // 4. k = concat(x_proj, slots_norm) @ wk^T + bk   (dual-source rows)
  gemm_f16_wmma<h16, h16><<<dim3(4, 320), 256, 0, stream>>>(
      xp, snb, 640, 512, aiW + (size_t)512 * 512, aiB + 512, kb, 40960, 512, 512);
  // 5. v
  gemm_f16_wmma<h16, h16><<<dim3(4, 320), 256, 0, stream>>>(
      xp, snb, 640, 512, aiW + (size_t)1024 * 512, aiB + 1024, vb, 40960, 512, 512);
  // 6. attention (flash, per b,h)
  flash_attn_kernel<<<256, 256, 0, stream>>>(qb, kb, vb, atn);
  // 7. attn_out = attn @ attn_out_w^T + b
  gemm_f16_wmma<h16, float><<<dim3(4, 64), 256, 0, stream>>>(
      atn, (const h16*)nullptr, 0, 0, aoW, aoB, aob, 8192, 512, 512);
  // 8. gate top2 (expert scale == 1 exactly; only idx/weights are outputs)
  gate_topk_kernel<<<64, 256, 0, stream>>>(aob, gW, gB, out);
  // 9. expert_out = attn_out @ expert_w^T + b
  gemm_f16_wmma<float, float><<<dim3(4, 64), 256, 0, stream>>>(
      aob, (const float*)nullptr, 0, 0, exW, exB, eob, 8192, 512, 512);
  // 10. slot gate + top8 (+ read top8)
  slot_topk_kernel<<<64, 256, 0, stream>>>(eob, sgW, sgB, rsc, sidx, out);
  // 11. gather
  gather_kernel<<<64, 256, 0, stream>>>(eob, st, sidx, geo, gst);
  // 12. upd_pre = gath_eo @ state_w^T + b
  gemm_f16_wmma<float, float><<<dim3(4, 4), 256, 0, stream>>>(
      geo, (const float*)nullptr, 0, 0, stW, stB, updb, 512, 512, 512);
  // 13. next_state = state, then scatter updated rows
  copy_state_kernel<<<4096, 256, 0, stream>>>(st, ns);
  scatter_kernel<<<64, 256, 0, stream>>>(gst, updb, sidx, out + OFF_SW, ns);
  // 14. output = next_state @ out_w^T + b   (M=64, K=65536)
  gemm_f16_wmma<float, float><<<dim3(4, 1), 256, 0, stream>>>(
      ns, (const float*)nullptr, 0, 0, oW, oB, out, 64, 512, 65536);
}
